// Point_RPE_MAP_Encoder_55722905698618
// MI455X (gfx1250) — matrix operand order repaired and checked by reading, not compiled, not run on hardware
//
#include <hip/hip_runtime.h>

// ---------------------------------------------------------------------------
// Point-RPE MAP encoder for MI455X (gfx1250, wave32, WMMA).
// All 128-wide GEMMs run through v_wmma_f32_16x16x32_f16 (f16 in, f32 acc),
// LayerNorm fused into the GEMM epilogue via LDS + wave32 shuffles.
// ---------------------------------------------------------------------------

typedef __attribute__((ext_vector_type(16))) _Float16 v16h;
typedef __attribute__((ext_vector_type(8)))  _Float16 v8h;
typedef __attribute__((ext_vector_type(8)))  float    v8f;

#define NPOINTS 200000
#define NLANES  10000
#define NEDGES  320000

#define CDIV(a,b) (((a)+(b)-1)/(b))

// ---------------- device helpers ----------------

__device__ __forceinline__ float wred_sum(float v) {
  #pragma unroll
  for (int m = 16; m >= 1; m >>= 1) v += __shfl_xor(v, m, 32);
  return v;
}

// monotonic float->uint key for atomicMax-based segment max
__device__ __forceinline__ unsigned f2key(float f) {
  unsigned u = __float_as_uint(f);
  return (u & 0x80000000u) ? ~u : (u | 0x80000000u);
}
// key 0 == "empty segment" -> 0.0 (torch_scatter / seg_max0 semantics)
__device__ __forceinline__ float key2f0(unsigned k) {
  if (k == 0u) return 0.0f;
  unsigned u = (k & 0x80000000u) ? (k & 0x7FFFFFFFu) : ~k;
  return __uint_as_float(u);
}

// ---------------- WMMA GEMM (+bias +residual +LayerNorm +ReLU) ----------------
// A: [M,K] f16 row-major (M % 16 == 0, K % 32 == 0)
// Wt: [N,K] f16 row-major (weights pre-transposed)
// one wave -> 16x16 output tile; workgroup (N/16 waves) -> full 16-row block,
// enabling row-wise LayerNorm in the epilogue.
template <int N>
__global__ void gemm_wmma_k(const _Float16* __restrict__ A, int K,
                            const _Float16* __restrict__ Wt,
                            const float* __restrict__ bias,
                            const float* __restrict__ res,
                            const float* __restrict__ ln_g,
                            const float* __restrict__ ln_b,
                            int do_relu,
                            float* __restrict__ out32,
                            _Float16* __restrict__ out16) {
  __shared__ float tile[16][N];
  const int lane = threadIdx.x & 31;
  const int wave = threadIdx.x >> 5;
  const int l15  = lane & 15;
  const int hi   = lane >> 4;        // lanes 16..31
  const int m0   = blockIdx.x * 16;
  const int n0   = wave * 16;

  // A fragment: lane holds row m0+l15.
  //   lanes 0-15 : K = k0+[0..15]
  //   lanes 16-31: K = k0+[16..31]
  // B fragment (Wt row n0+l15): K = k0 + (hi?16:0) + [0..15] contiguous.
  const _Float16* Ap = A  + (size_t)(m0 + l15) * K + (hi ? 16 : 0);
  const _Float16* Bp = Wt + (size_t)(n0 + l15) * K + (hi ? 16 : 0);

  v8f acc = {};
  for (int k0 = 0; k0 < K; k0 += 32) {
    v8h alo = *(const v8h*)(Ap + k0);
    v8h ahi = *(const v8h*)(Ap + k0 + 8);
    v16h a;
    #pragma unroll
    for (int i = 0; i < 8; ++i) { a[i] = alo[i]; a[i + 8] = ahi[i]; }
    v16h b = *(const v16h*)(Bp + k0);
    acc = __builtin_amdgcn_wmma_f32_16x16x32_f16(
        /*neg_a=*/false, a, /*neg_b=*/false, b,
        /*c_mod=*/(short)0, acc, /*reuse_a=*/false, /*reuse_b=*/false);
  }
  // D layout: acc[j] -> row m0 + j + hi*8, col n0 + l15
  #pragma unroll
  for (int j = 0; j < 8; ++j) tile[j + hi * 8][n0 + l15] = acc[j];
  __syncthreads();

  constexpr int NW  = N / 16;   // waves
  constexpr int RPW = 16 / NW;  // rows per wave
  constexpr int CPL = N / 32;   // cols per lane
  for (int rr = 0; rr < RPW; ++rr) {
    const int r = wave * RPW + rr;
    const size_t row = (size_t)(m0 + r);
    float vals[CPL];
    float s1 = 0.f, s2 = 0.f;
    #pragma unroll
    for (int c = 0; c < CPL; ++c) {
      const int col = lane * CPL + c;
      float t = tile[r][col];
      if (bias) t += bias[col];
      if (res)  t += res[row * N + col];
      vals[c] = t; s1 += t; s2 += t * t;
    }
    if (ln_g) {
      s1 = wred_sum(s1); s2 = wred_sum(s2);
      const float mean = s1 * (1.0f / N);
      const float var  = s2 * (1.0f / N) - mean * mean;   // biased var
      const float rstd = rsqrtf(var + 1e-5f);
      #pragma unroll
      for (int c = 0; c < CPL; ++c) {
        const int col = lane * CPL + c;
        vals[c] = (vals[c] - mean) * rstd * ln_g[col] + ln_b[col];
      }
    }
    if (do_relu) {
      #pragma unroll
      for (int c = 0; c < CPL; ++c) vals[c] = fmaxf(vals[c], 0.0f);
    }
    #pragma unroll
    for (int c = 0; c < CPL; ++c) {
      const int col = lane * CPL + c;
      if (out32) out32[row * N + col] = vals[c];
      if (out16) out16[row * N + col] = (_Float16)vals[c];
    }
  }
}

// ---------------- small-K row kernel: relu(LN(x @ W + b)) ----------------
// one wave per output row of 128; K <= 16 (proj K=10, rpe K=8)
__global__ void smallk_ln_relu_k(const float* __restrict__ X, int K, int M,
                                 const float* __restrict__ W,    // [K,128]
                                 const float* __restrict__ bias, // [128]
                                 const float* __restrict__ g,
                                 const float* __restrict__ b,
                                 float* __restrict__ out32,
                                 _Float16* __restrict__ out16) {
  const int lane = threadIdx.x & 31;
  const int wave = threadIdx.x >> 5;
  const int row  = blockIdx.x * 8 + wave;
  if (row >= M) return;
  float acc[4];
  #pragma unroll
  for (int c = 0; c < 4; ++c) acc[c] = bias[lane * 4 + c];
  for (int k = 0; k < K; ++k) {
    const float xv = X[(size_t)row * K + k];
    #pragma unroll
    for (int c = 0; c < 4; ++c) acc[c] += xv * W[k * 128 + lane * 4 + c];
  }
  float s1 = 0.f, s2 = 0.f;
  #pragma unroll
  for (int c = 0; c < 4; ++c) { s1 += acc[c]; s2 += acc[c] * acc[c]; }
  s1 = wred_sum(s1); s2 = wred_sum(s2);
  const float mean = s1 * (1.0f / 128.0f);
  const float rstd = rsqrtf(s2 * (1.0f / 128.0f) - mean * mean + 1e-5f);
  #pragma unroll
  for (int c = 0; c < 4; ++c) {
    const int col = lane * 4 + c;
    float v = (acc[c] - mean) * rstd * g[col] + b[col];
    v = fmaxf(v, 0.0f);
    out32[(size_t)row * 128 + col] = v;
    out16[(size_t)row * 128 + col] = (_Float16)v;
  }
}

// ---------------- residual + LayerNorm: out = LN(A + B) ----------------
__global__ void ln_residual_k(const float* __restrict__ A,
                              const float* __restrict__ B, int M,
                              const float* __restrict__ g,
                              const float* __restrict__ b,
                              float* __restrict__ out32,
                              _Float16* __restrict__ out16) {
  const int lane = threadIdx.x & 31;
  const int wave = threadIdx.x >> 5;
  const int row  = blockIdx.x * 8 + wave;
  if (row >= M) return;
  float vals[4];
  float s1 = 0.f, s2 = 0.f;
  #pragma unroll
  for (int c = 0; c < 4; ++c) {
    const size_t i = (size_t)row * 128 + lane * 4 + c;
    float t = A[i] + B[i];
    vals[c] = t; s1 += t; s2 += t * t;
  }
  s1 = wred_sum(s1); s2 = wred_sum(s2);
  const float mean = s1 * (1.0f / 128.0f);
  const float rstd = rsqrtf(s2 * (1.0f / 128.0f) - mean * mean + 1e-5f);
  #pragma unroll
  for (int c = 0; c < 4; ++c) {
    const int col = lane * 4 + c;
    float v = (vals[c] - mean) * rstd * g[col] + b[col];
    out32[(size_t)row * 128 + col] = v;
    if (out16) out16[(size_t)row * 128 + col] = (_Float16)v;
  }
}

// ---------------- utility kernels ----------------

__global__ void fill_u32_k(unsigned* p, unsigned v, size_t n) {
  size_t i = (size_t)blockIdx.x * 256 + threadIdx.x;
  if (i < n) p[i] = v;
}
__global__ void fill_f32_k(float* p, float v, size_t n) {
  size_t i = (size_t)blockIdx.x * 256 + threadIdx.x;
  if (i < n) p[i] = v;
}
__global__ void copy_f32_k(float* dst, const float* src, size_t n) {
  size_t i = (size_t)blockIdx.x * 256 + threadIdx.x;
  if (i < n) dst[i] = src[i];
}
// W[K,N] f32 -> Wt[N,K] f16
__global__ void wt_f16_k(const float* __restrict__ W, _Float16* __restrict__ Wt,
                         int K, int N) {
  int idx = blockIdx.x * 256 + threadIdx.x;
  if (idx >= K * N) return;
  int n = idx / K, k = idx - n * K;
  Wt[idx] = (_Float16)W[(size_t)k * N + n];
}

// segment max via transformed-uint atomicMax over [M,128]
__global__ void segmax_scatter_k(const float* __restrict__ X,
                                 const int* __restrict__ seg,
                                 unsigned* __restrict__ keys, int M) {
  size_t i = (size_t)blockIdx.x * 256 + threadIdx.x;
  if (i >= (size_t)M * 128) return;
  int row = (int)(i >> 7), col = (int)(i & 127);
  atomicMax(&keys[(size_t)seg[row] * 128 + col], f2key(X[i]));
}
__global__ void seg_decode_k(const unsigned* __restrict__ keys,
                             float* __restrict__ out32,
                             _Float16* __restrict__ out16, size_t n) {
  size_t i = (size_t)blockIdx.x * 256 + threadIdx.x;
  if (i >= n) return;
  float v = key2f0(keys[i]);
  out32[i] = v;
  if (out16) out16[i] = (_Float16)v;
}

// xa = concat(x, mx[seg]) as f16 [M,256]
__global__ void build_xa_k(const float* __restrict__ X,
                           const float* __restrict__ MX,
                           const int* __restrict__ seg,
                           _Float16* __restrict__ xa, int M) {
  size_t i = (size_t)blockIdx.x * 256 + threadIdx.x;
  if (i >= (size_t)M * 128) return;
  int row = (int)(i >> 7), col = (int)(i & 127);
  xa[(size_t)row * 256 + col]       = (_Float16)X[i];
  xa[(size_t)row * 256 + 128 + col] = (_Float16)MX[(size_t)seg[row] * 128 + col];
}

// mem_in = concat(x[dst], x[src], attr) as f16 [E,384]
__global__ void build_memin_k(const _Float16* __restrict__ lane16,
                              const _Float16* __restrict__ attr16,
                              const int* __restrict__ src,
                              const int* __restrict__ dst,
                              _Float16* __restrict__ mi, int E) {
  size_t i = (size_t)blockIdx.x * 256 + threadIdx.x;
  if (i >= (size_t)E * 128) return;
  int e = (int)(i >> 7), c = (int)(i & 127);
  mi[(size_t)e * 384 + c]       = lane16[(size_t)dst[e] * 128 + c];
  mi[(size_t)e * 384 + 128 + c] = lane16[(size_t)src[e] * 128 + c];
  mi[(size_t)e * 384 + 256 + c] = attr16[i];
}

// per-edge per-head logits = dot(q[dst], k)/sqrt(16), + segment max
__global__ void logits_k(const float* __restrict__ Q,
                         const float* __restrict__ Kf,
                         const int* __restrict__ dst,
                         float* __restrict__ lg,
                         unsigned* __restrict__ mkey, int E) {
  int idx = blockIdx.x * 256 + threadIdx.x;
  if (idx >= E * 8) return;
  int e = idx >> 3, h = idx & 7;
  const int d0 = dst[e];
  const float* q  = Q  + (size_t)d0 * 128 + h * 16;
  const float* kk = Kf + (size_t)e  * 128 + h * 16;
  float s = 0.f;
  #pragma unroll
  for (int d = 0; d < 16; ++d) s += q[d] * kk[d];
  s *= 0.25f;  // 1/sqrt(16)
  lg[idx] = s;
  atomicMax(&mkey[d0 * 8 + h], f2key(s));
}

__global__ void expsum_k(float* __restrict__ lg,
                         const unsigned* __restrict__ mkey,
                         const int* __restrict__ dst,
                         float* __restrict__ ssum, int E) {
  int idx = blockIdx.x * 256 + threadIdx.x;
  if (idx >= E * 8) return;
  int e = idx >> 3, h = idx & 7;
  const float m = key2f0(mkey[dst[e] * 8 + h]);
  const float w = __expf(lg[idx] - m);
  lg[idx] = w;
  atomicAdd(&ssum[dst[e] * 8 + h], w);
}

// wv = (attn * v) as f16, head-major channels (matches reshape(-1, D))
__global__ void wv_k(const float* __restrict__ lg,
                     const float* __restrict__ ssum,
                     const float* __restrict__ V,
                     const int* __restrict__ dst,
                     _Float16* __restrict__ wv, int E) {
  int idx = blockIdx.x * 256 + threadIdx.x;
  if (idx >= E * 8) return;
  int e = idx >> 3, h = idx & 7;
  const float a = lg[idx] / (ssum[dst[e] * 8 + h] + 1e-16f);
  #pragma unroll
  for (int d = 0; d < 16; ++d) {
    const size_t i = (size_t)e * 128 + h * 16 + d;
    wv[i] = (_Float16)(a * V[i]);
  }
}

// segment sum of out_e into aggr[dst]
__global__ void scatter_add_k(const float* __restrict__ X,
                              const int* __restrict__ dst,
                              float* __restrict__ out, int E) {
  size_t i = (size_t)blockIdx.x * 256 + threadIdx.x;
  if (i >= (size_t)E * 128) return;
  int e = (int)(i >> 7), c = (int)(i & 127);
  atomicAdd(&out[(size_t)dst[e] * 128 + c], X[i]);
}

// ---------------- host side ----------------

extern "C" void kernel_launch(void* const* d_in, const int* in_sizes, int n_in,
                              void* d_out, int out_size, void* d_ws, size_t ws_size,
                              hipStream_t stream) {
  (void)in_sizes; (void)n_in; (void)out_size; (void)ws_size;
  const int NP = NPOINTS, NL = NLANES, NE = NEDGES;

  auto F = [&](int i) { return (const float*)d_in[i]; };
  const float* node_feats = F(0);
  const int*   seg  = (const int*)d_in[1];
  const int*   e2   = (const int*)d_in[2];
  const int*   esrc = e2;           // edges[0]
  const int*   edst = e2 + NE;      // edges[1]
  const float* rpes = F(3);

  // param leaf indices (setup_inputs insertion order, recursive):
  // 4..7: proj_l.w, proj_l.b, proj_n.g, proj_n.b
  // 8..11: rpe_l.w, rpe_l.b, rpe_n.g, rpe_n.b
  // 12..29: point_aggr, 30..47: lane_aggr  (per block:
  //   +0 fc1_l1.w +1 .b  +2 fc1_n1.g +3 .b  +4 fc1_l2.w +5 .b  +6 fc1_n2.g +7 .b
  //   +8 fc2_l1.w +9 .b  +10 fc2_n1.g +11 .b  +12 fc2_l2.w +13 .b
  //   +14 fc2_n2.g +15 .b  +16 norm.g +17 .b)
  // layers l=0,1 at 48+22*l (per layer:
  //   +0 mem_l.w +1 .b  +2 mem_n.g +3 .b  +4 q.w +5 k.w +6 v.w +7 o.w
  //   +8 eu_l.w +9 .b  +10 eu_n.g +11 .b  +12 edge_norm.g +13 .b
  //   +14 ffn1.w +15 .b  +16 ffn2.w +17 .b  +18 norm1.g +19 .b +20 norm2.g +21 .b)
  const int PB = 12, LB = 30;
  auto LY = [&](int l, int off) { return 48 + 22 * l + off; };

  // ---- workspace carve (256B aligned) ----
  char* cur = (char*)d_ws;
  auto alloc = [&](size_t bytes) -> void* {
    void* p = (void*)cur;
    cur += (bytes + 255) & ~(size_t)255;
    return p;
  };
  float*     P0  = (float*)alloc((size_t)NP * 128 * 4);
  float*     P1  = (float*)alloc((size_t)NP * 128 * 4);
  float*     P2  = (float*)alloc((size_t)NP * 128 * 4);
  _Float16*  PH0 = (_Float16*)alloc((size_t)NP * 256 * 2);
  _Float16*  PH1 = (_Float16*)alloc((size_t)NP * 256 * 2);
  float*     E0  = (float*)alloc((size_t)NE * 128 * 4);   // attr
  float*     E1  = (float*)alloc((size_t)NE * 128 * 4);   // k
  float*     E2  = (float*)alloc((size_t)NE * 128 * 4);   // v
  float*     E3  = (float*)alloc((size_t)NE * 128 * 4);   // delta / out_e
  _Float16*  EH0 = (_Float16*)alloc((size_t)NE * 384 * 2);// mem_in
  _Float16*  EH1 = (_Float16*)alloc((size_t)NE * 128 * 2);// mem16 / wv16
  _Float16*  EH3 = (_Float16*)alloc((size_t)NE * 128 * 2);// attr16
  float*     ELG = (float*)alloc((size_t)NE * 8 * 4);     // logits / exp
  unsigned*  MXK = (unsigned*)alloc((size_t)NL * 128 * 4);
  float*     MX  = (float*)alloc((size_t)NL * 128 * 4);
  float*     L0  = (float*)alloc((size_t)NL * 128 * 4);   // lane feats
  float*     L1  = (float*)alloc((size_t)NL * 128 * 4);   // aggr
  float*     L2  = (float*)alloc((size_t)NL * 128 * 4);   // q
  _Float16*  LH0 = (_Float16*)alloc((size_t)NL * 128 * 2);
  _Float16*  LH1 = (_Float16*)alloc((size_t)NL * 256 * 2);
  unsigned*  MKEY = (unsigned*)alloc((size_t)NL * 8 * 4);
  float*     SSUM = (float*)alloc((size_t)NL * 8 * 4);
  // f16 transposed weights
  auto allocW = [&](size_t n) { return (_Float16*)alloc(n * 2); };
  _Float16 *wp11 = allocW(128 * 128), *wp12 = allocW(128 * 128);
  _Float16 *wp21 = allocW(128 * 256), *wp22 = allocW(128 * 128);
  _Float16 *wl11 = allocW(128 * 128), *wl12 = allocW(128 * 128);
  _Float16 *wl21 = allocW(128 * 256), *wl22 = allocW(128 * 128);
  _Float16 *wmem[2], *wq[2], *wk[2], *wv[2], *wo[2], *weu[2], *wf1[2], *wf2[2];
  for (int l = 0; l < 2; ++l) {
    wmem[l] = allocW(128 * 384);
    wq[l] = allocW(128 * 128); wk[l] = allocW(128 * 128);
    wv[l] = allocW(128 * 128); wo[l] = allocW(128 * 128);
    weu[l] = allocW(128 * 128);
    wf1[l] = allocW(256 * 128);  // Wt[256][128]
    wf2[l] = allocW(128 * 256);  // Wt[128][256]
  }

  // ---- weight prep: f32 [K,N] -> f16 [N,K] ----
  auto prep = [&](int idx, int K, int N, _Float16* dst) {
    wt_f16_k<<<CDIV(K * N, 256), 256, 0, stream>>>(F(idx), dst, K, N);
  };
  prep(PB + 0, 128, 128, wp11); prep(PB + 4, 128, 128, wp12);
  prep(PB + 8, 256, 128, wp21); prep(PB + 12, 128, 128, wp22);
  prep(LB + 0, 128, 128, wl11); prep(LB + 4, 128, 128, wl12);
  prep(LB + 8, 256, 128, wl21); prep(LB + 12, 128, 128, wl22);
  for (int l = 0; l < 2; ++l) {
    prep(LY(l, 0), 384, 128, wmem[l]);
    prep(LY(l, 4), 128, 128, wq[l]);  prep(LY(l, 5), 128, 128, wk[l]);
    prep(LY(l, 6), 128, 128, wv[l]);  prep(LY(l, 7), 128, 128, wo[l]);
    prep(LY(l, 8), 128, 128, weu[l]);
    prep(LY(l, 14), 128, 256, wf1[l]);
    prep(LY(l, 16), 256, 128, wf2[l]);
  }

  auto g128 = [&](const _Float16* A, int K, const _Float16* Wt,
                  const float* bias, const float* res,
                  const float* g, const float* b, int relu,
                  float* o32, _Float16* o16, int M) {
    gemm_wmma_k<128><<<M / 16, 256, 0, stream>>>(A, K, Wt, bias, res, g, b,
                                                 relu, o32, o16);
  };

  // ---- proj: x = relu(LN(node_feats @ proj_w + b)) ----
  smallk_ln_relu_k<<<CDIV(NP, 8), 256, 0, stream>>>(
      node_feats, 10, NP, F(4), F(5), F(6), F(7), P0, PH0);

  // ---- agg block (point_aggr then lane_aggr), NP rows each ----
  auto agg_block = [&](int B, float* xin32, _Float16* xin16,
                       float* t1_32, _Float16* t1_16,
                       float* x32, _Float16* x16_dst, float* out32) {
    // x = relu(LN(fc1_l2(relu(LN(fc1_l1(x_inp))))))
    g128(xin16, 128, (B == PB) ? wp11 : wl11, F(B + 1), nullptr,
         F(B + 2), F(B + 3), 1, t1_32, t1_16, NP);
    g128(t1_16, 128, (B == PB) ? wp12 : wl12, F(B + 5), nullptr,
         F(B + 6), F(B + 7), 1, x32, x16_dst, NP);
    // mx = seg_max0(x)
    fill_u32_k<<<CDIV(NL * 128, 256), 256, 0, stream>>>(MXK, 0u, (size_t)NL * 128);
    segmax_scatter_k<<<CDIV(NP * 128, 256), 256, 0, stream>>>(x32, seg, MXK, NP);
    seg_decode_k<<<CDIV(NL * 128, 256), 256, 0, stream>>>(MXK, MX, nullptr,
                                                          (size_t)NL * 128);
    // xa = concat(x, mx[seg]) f16 [NP,256]
    build_xa_k<<<CDIV(NP * 128, 256), 256, 0, stream>>>(x32, MX, seg, PH1, NP);
    // h = relu(LN(fc2_l2(relu(LN(fc2_l1(xa))))))
    g128(PH1, 256, (B == PB) ? wp21 : wl21, F(B + 9), nullptr,
         F(B + 10), F(B + 11), 1, nullptr, PH0, NP);
    g128(PH0, 128, (B == PB) ? wp22 : wl22, F(B + 13), nullptr,
         F(B + 14), F(B + 15), 1, x32, nullptr, NP);   // reuse x32 as h
    // out = LN(x_inp + h)
    ln_residual_k<<<CDIV(NP, 8), 256, 0, stream>>>(
        xin32, x32, NP, F(B + 16), F(B + 17), out32,
        (B == PB) ? PH0 : nullptr);  // point block: out16 feeds lane block
  };
  // point_aggr: x_inp = P0/PH0 -> out P1 (+PH0 f16)
  agg_block(PB, P0, PH0, P1, PH1, P2, PH0, P1);
  // lane_aggr: x_inp = P1/PH0 -> out P0, then lane = seg_max0(out)
  agg_block(LB, P1, PH0, P0, PH1, P2, PH0, P0);
  fill_u32_k<<<CDIV(NL * 128, 256), 256, 0, stream>>>(MXK, 0u, (size_t)NL * 128);
  segmax_scatter_k<<<CDIV(NP * 128, 256), 256, 0, stream>>>(P0, seg, MXK, NP);
  seg_decode_k<<<CDIV(NL * 128, 256), 256, 0, stream>>>(MXK, L0, LH0,
                                                        (size_t)NL * 128);

  // ---- attr = relu(LN(rpe(l2l_fused_rpes))) ----
  smallk_ln_relu_k<<<CDIV(NE, 8), 256, 0, stream>>>(
      rpes, 8, NE, F(8), F(9), F(10), F(11), E0, EH3);

  // ---- GAT layers ----
  for (int l = 0; l < 2; ++l) {
    // mem_in = concat(x[dst], x[src], attr) f16 [NE,384]
    build_memin_k<<<CDIV(NE * 128, 256), 256, 0, stream>>>(LH0, EH3, esrc, edst,
                                                           EH0, NE);
    // mem = relu(LN(mem_l(mem_in)))  (f16 only)
    g128(EH0, 384, wmem[l], F(LY(l, 1)), nullptr, F(LY(l, 2)), F(LY(l, 3)), 1,
         nullptr, EH1, NE);
    // delta = relu(LN(eu_l(mem)))
    g128(EH1, 128, weu[l], F(LY(l, 9)), nullptr, F(LY(l, 10)), F(LY(l, 11)), 1,
         E3, nullptr, NE);
    // attr = LN(attr + delta)
    ln_residual_k<<<CDIV(NE, 8), 256, 0, stream>>>(
        E0, E3, NE, F(LY(l, 12)), F(LY(l, 13)), E0, EH3);
    // q (lanes), k, v (edges) -- no bias, no LN
    g128(LH0, 128, wq[l], nullptr, nullptr, nullptr, nullptr, 0, L2, nullptr, NL);
    g128(EH1, 128, wk[l], nullptr, nullptr, nullptr, nullptr, 0, E1, nullptr, NE);
    g128(EH1, 128, wv[l], nullptr, nullptr, nullptr, nullptr, 0, E2, nullptr, NE);
    // scatter softmax over dst
    fill_u32_k<<<CDIV(NL * 8, 256), 256, 0, stream>>>(MKEY, 0u, (size_t)NL * 8);
    fill_f32_k<<<CDIV(NL * 8, 256), 256, 0, stream>>>(SSUM, 0.0f, (size_t)NL * 8);
    logits_k<<<CDIV(NE * 8, 256), 256, 0, stream>>>(L2, E1, edst, ELG, MKEY, NE);
    expsum_k<<<CDIV(NE * 8, 256), 256, 0, stream>>>(ELG, MKEY, edst, SSUM, NE);
    wv_k<<<CDIV(NE * 8, 256), 256, 0, stream>>>(ELG, SSUM, E2, edst, EH1, NE);
    // out_e = o(attn*v); aggr = segment_sum(out_e, dst)
    g128(EH1, 128, wo[l], nullptr, nullptr, nullptr, nullptr, 0, E3, nullptr, NE);
    fill_f32_k<<<CDIV(NL * 128, 256), 256, 0, stream>>>(L1, 0.0f,
                                                        (size_t)NL * 128);
    scatter_add_k<<<CDIV(NE * 128, 256), 256, 0, stream>>>(E3, edst, L1, NE);
    // x = LN(x + aggr)
    ln_residual_k<<<CDIV(NL, 8), 256, 0, stream>>>(
        L0, L1, NL, F(LY(l, 18)), F(LY(l, 19)), L0, LH0);
    // ffn: x = LN(x + ffn2(relu(ffn1(x))))
    gemm_wmma_k<256><<<NL / 16, 512, 0, stream>>>(
        LH0, 128, wf1[l], F(LY(l, 15)), nullptr, nullptr, nullptr, 1,
        nullptr, LH1);
    g128(LH1, 256, wf2[l], F(LY(l, 17)), L0, F(LY(l, 20)), F(LY(l, 21)), 0,
         L0, LH0, NL);
  }

  // ---- outputs: [lane (NL*128), attr (NE*128)] ----
  float* out = (float*)d_out;
  copy_f32_k<<<CDIV(NL * 128, 256), 256, 0, stream>>>(out, L0, (size_t)NL * 128);
  copy_f32_k<<<CDIV(NE * 128, 256), 256, 0, stream>>>(out + (size_t)NL * 128, E0,
                                                      (size_t)NE * 128);
}
